// CrossAttention_39256001085949
// MI455X (gfx1250) — compile-verified, matrix operand order checked by reading, not run on hardware
//
#include <hip/hip_runtime.h>
#include <hip/hip_bf16.h>
#include <math.h>

typedef _Float16 f16;
typedef __attribute__((ext_vector_type(16))) _Float16 v16h;
typedef __attribute__((ext_vector_type(8)))  _Float16 v8h;
typedef __attribute__((ext_vector_type(8)))  float    v8f;

constexpr int Bc = 2, Nc = 2048, Cc = 1024, Hc = 16, Dc = 64;
constexpr int Kdim = Cc;   // K = 1024 for every projection GEMM

static __device__ inline v16h cat16(v8h a, v8h b) {
  v16h r;
#pragma unroll
  for (int i = 0; i < 8; ++i) { r[i] = a[i]; r[i + 8] = b[i]; }
  return r;
}

static __device__ inline v8f wmma_f16(v16h a, v16h b, v8f c) {
  // 8 args: (neg_a, A, neg_b, B, c_mod, C, reuse_a, reuse_b)
  return __builtin_amdgcn_wmma_f32_16x16x32_f16(false, a, false, b, (short)0, c,
                                                false, false);
}

// CDNA5 async copy: 16B per lane, global -> LDS, tracked by ASYNCcnt.
static __device__ inline void async_ld_b128(unsigned int lds_off,
                                            const void* gaddr) {
  asm volatile("global_load_async_to_lds_b128 %0, %1, off"
               :: "v"(lds_off), "v"((unsigned long long)(uintptr_t)gaddr)
               : "memory");
}

// ---------------------------------------------------------------- convert ---
__global__ void __launch_bounds__(256)
cvt_f32_f16(const float* __restrict__ s, f16* __restrict__ d, int n) {
  int i = blockIdx.x * blockDim.x + threadIdx.x;
  int stride = gridDim.x * blockDim.x;
  for (; i < n; i += stride) d[i] = (f16)s[i];
}

// -------------------------------------------------------- generic A * W^T ---
// A: [4096, 1024] f16 row-major. W: [Ncols, 1024] f16 row-major.
// mode 0: q   -> out16 = qh  [B,H,N,D]
// mode 1: kv  -> out16 = kh  [B,H,N,D],  out16b = v^T [B,H,D,N]
// mode 2: out -> out32 = [B,N,C] f32 + bias
__global__ void __launch_bounds__(128)
gemm_awt(const f16* __restrict__ A, const f16* __restrict__ W,
         const float* __restrict__ bias, f16* __restrict__ out16,
         f16* __restrict__ out16b, float* __restrict__ out32, int mode) {
  const int lane = threadIdx.x & 31;
  const int wave = threadIdx.x >> 5;
  const int l16  = lane & 15;
  const int hi   = lane >> 4;
  const int kh8  = hi * 8;                    // K sub-offset per ISA A/B layout
  const int m0   = blockIdx.x * 64 + wave * 16;
  const int n0   = blockIdx.y * 64;

  v8f acc[4];
#pragma unroll
  for (int t = 0; t < 4; ++t)
#pragma unroll
    for (int i = 0; i < 8; ++i) acc[t][i] = 0.0f;

  const f16* Ap = A + (size_t)(m0 + l16) * Kdim;
  for (int k0 = 0; k0 < Kdim; k0 += 32) {
    v16h afrag = cat16(*(const v8h*)(Ap + k0 + kh8),
                       *(const v8h*)(Ap + k0 + kh8 + 16));
#pragma unroll
    for (int t = 0; t < 4; ++t) {
      const f16* Wp = W + (size_t)(n0 + t * 16 + l16) * Kdim + k0 + kh8;
      v16h bfrag = cat16(*(const v8h*)Wp, *(const v8h*)(Wp + 16));
      acc[t] = wmma_f16(afrag, bfrag, acc[t]);
    }
  }

#pragma unroll
  for (int t = 0; t < 4; ++t) {
#pragma unroll
    for (int r = 0; r < 8; ++r) {
      const int m = m0 + r + 8 * hi;          // C-layout: lanes16-31 hold M=8+r
      const int c = n0 + t * 16 + l16;
      const int b = m >> 11;                  // m / Nc
      const int n = m & (Nc - 1);
      const float val = acc[t][r];
      if (mode == 0) {
        const int h = c >> 6, d = c & 63;
        out16[((size_t)((b * Hc + h) * Nc + n)) * Dc + d] = (f16)val;
      } else if (mode == 1) {
        if (c < Cc) {
          const int h = c >> 6, d = c & 63;
          out16[((size_t)((b * Hc + h) * Nc + n)) * Dc + d] = (f16)val;
        } else {
          const int c2 = c - Cc, h = c2 >> 6, d = c2 & 63;
          out16b[((size_t)((b * Hc + h) * Dc + d)) * Nc + n] = (f16)val;
        }
      } else {
        out32[(size_t)m * Cc + c] = val + bias[c];
      }
    }
  }
}

// ----------------------------------------------------------- flash attention -
// grid: (N/64, B*H), block 128 (4 waves, 16 query rows per wave).
// K/V staged cooperatively via CDNA5 async global->LDS copies, double-buffered.
__global__ void __launch_bounds__(128)
attn_kernel(const f16* __restrict__ qh, const f16* __restrict__ kh,
            const f16* __restrict__ vt, f16* __restrict__ oh) {
  // padded rows: 72 f16 (144B = 36 banks) / 40 f16 (80B = 20 banks), 16B-aligned
  __shared__ __align__(16) f16 kbuf[2][32][72];   // keys j..j+31, d 0..63
  __shared__ __align__(16) f16 vbuf[2][64][40];   // v^T rows d 0..63, keys 0..31
  __shared__ __align__(16) f16 pbuf[4][16][40];   // per-wave P transpose tile

  const int tid  = threadIdx.x;
  const int lane = tid & 31;
  const int wave = tid >> 5;
  const int l16  = lane & 15;
  const int hi   = lane >> 4;
  const int kh8  = hi * 8;
  const int bh   = blockIdx.y;                      // b*H + h
  const int n0   = blockIdx.x * 64 + wave * 16;     // query rows of this wave

  const f16* kbase = kh + (size_t)bh * Nc * Dc;
  const f16* vbase = vt + (size_t)bh * Dc * Nc;

  // cooperative async stage of one 32-key K tile + V^T tile (4 instrs/wave)
  auto stage = [&](int bufi, int jj) {
#pragma unroll
    for (int u = 0; u < 2; ++u) {
      const int unit = tid + u * 128;               // 256 chunks of 8 f16
      const int row = unit >> 3, ch = (unit & 7) * 8;
      async_ld_b128((unsigned int)(uintptr_t)&kbuf[bufi][row][ch],
                    kbase + (size_t)(jj + row) * Dc + ch);
    }
#pragma unroll
    for (int u = 0; u < 2; ++u) {
      const int unit = tid + u * 128;
      const int row = unit >> 2, ch = (unit & 3) * 8;
      async_ld_b128((unsigned int)(uintptr_t)&vbuf[bufi][row][ch],
                    vbase + (size_t)row * Nc + jj + ch);
    }
  };

  // q fragments for K = D = 64 (two K=32 fragments), pre-scaled by 1/sqrt(D)
  const f16* qp = qh + ((size_t)bh * Nc + n0 + l16) * Dc;
  v16h qf[2];
#pragma unroll
  for (int kk = 0; kk < 2; ++kk) {
    v8h a0 = *(const v8h*)(qp + kk * 32 + kh8);
    v8h a1 = *(const v8h*)(qp + kk * 32 + kh8 + 16);
#pragma unroll
    for (int i = 0; i < 8; ++i) { a0[i] *= (f16)0.125f; a1[i] *= (f16)0.125f; }
    qf[kk] = cat16(a0, a1);
  }

  v8f accO[4];
#pragma unroll
  for (int t = 0; t < 4; ++t)
#pragma unroll
    for (int i = 0; i < 8; ++i) accO[t][i] = 0.0f;
  float mrow[8], lrow[8];
#pragma unroll
  for (int r = 0; r < 8; ++r) { mrow[r] = -INFINITY; lrow[r] = 0.0f; }

  stage(0, 0);

  for (int j = 0, it = 0; j < Nc; j += 32, ++it) {
    const int cur = it & 1;
    if (j + 32 < Nc) {
      stage(cur ^ 1, j + 32);
      asm volatile("s_wait_asynccnt 0x4" ::: "memory");  // current tile done
    } else {
      asm volatile("s_wait_asynccnt 0x0" ::: "memory");
    }
    __syncthreads();                                     // tile visible to WG

    // --- S = (q/8) * k^T : two 16x16 tiles covering 32 keys (from LDS) ---
    v8f s[2];
#pragma unroll
    for (int tt = 0; tt < 2; ++tt) {
      const f16* kp = &kbuf[cur][tt * 16 + l16][0];
      v16h b0 = cat16(*(const v8h*)(kp + kh8), *(const v8h*)(kp + kh8 + 16));
      v16h b1 = cat16(*(const v8h*)(kp + 32 + kh8),
                      *(const v8h*)(kp + 32 + kh8 + 16));
      v8f z;
#pragma unroll
      for (int i = 0; i < 8; ++i) z[i] = 0.0f;
      z = wmma_f16(qf[0], b0, z);
      s[tt] = wmma_f16(qf[1], b1, z);
    }

    // --- online softmax (row = slot r within each 16-lane half) ---
    float alpha[8];
#pragma unroll
    for (int r = 0; r < 8; ++r) {
      float t = fmaxf(s[0][r], s[1][r]);
      t = fmaxf(t, __shfl_xor(t, 1));
      t = fmaxf(t, __shfl_xor(t, 2));
      t = fmaxf(t, __shfl_xor(t, 4));
      t = fmaxf(t, __shfl_xor(t, 8));
      const float mnew = fmaxf(mrow[r], t);
      alpha[r] = __expf(mrow[r] - mnew);
      mrow[r] = mnew;
      const float p0 = __expf(s[0][r] - mnew);
      const float p1 = __expf(s[1][r] - mnew);
      s[0][r] = p0; s[1][r] = p1;
      float ps = p0 + p1;
      ps += __shfl_xor(ps, 1);
      ps += __shfl_xor(ps, 2);
      ps += __shfl_xor(ps, 4);
      ps += __shfl_xor(ps, 8);
      lrow[r] = lrow[r] * alpha[r] + ps;
    }
#pragma unroll
    for (int t = 0; t < 4; ++t)
#pragma unroll
      for (int r = 0; r < 8; ++r) accO[t][r] *= alpha[r];

    // --- C-layout P -> A-fragment layout via per-wave LDS tile ---
#pragma unroll
    for (int r = 0; r < 8; ++r) {
      pbuf[wave][r + 8 * hi][l16]      = (f16)s[0][r];
      pbuf[wave][r + 8 * hi][16 + l16] = (f16)s[1][r];
    }
    v16h pf;
    {
      // intra-wave LDS ordering: same-wave DS ops complete in order
      const f16* pp = &pbuf[wave][l16][0];
      pf = cat16(*(const v8h*)(pp + kh8), *(const v8h*)(pp + kh8 + 16));
    }

    // --- O += P * V (v^T tile in LDS -> contiguous B-fragment loads) ---
#pragma unroll
    for (int t = 0; t < 4; ++t) {
      const f16* vp = &vbuf[cur][t * 16 + l16][0];
      v16h vf = cat16(*(const v8h*)(vp + kh8), *(const v8h*)(vp + kh8 + 16));
      accO[t] = wmma_f16(pf, vf, accO[t]);
    }
    __syncthreads();   // whole WG done reading `cur` before it is re-staged
  }

  // normalize + store o as f16 [B,N,C]
  const int b = bh / Hc, h = bh % Hc;
  float linv[8];
#pragma unroll
  for (int r = 0; r < 8; ++r) linv[r] = 1.0f / lrow[r];
#pragma unroll
  for (int t = 0; t < 4; ++t)
#pragma unroll
    for (int r = 0; r < 8; ++r) {
      const size_t idx =
          ((size_t)(b * Nc + n0 + r + 8 * hi)) * Cc + h * Dc + t * 16 + l16;
      oh[idx] = (f16)(accO[t][r] * linv[r]);
    }
}

// ------------------------------------------------------------------ launch ---
extern "C" void kernel_launch(void* const* d_in, const int* in_sizes, int n_in,
                              void* d_out, int out_size, void* d_ws,
                              size_t ws_size, hipStream_t stream) {
  const float* x   = (const float*)d_in[0];
  const float* y   = (const float*)d_in[1];
  const float* Wq  = (const float*)d_in[2];
  const float* Wkv = (const float*)d_in[3];
  const float* Wp  = (const float*)d_in[4];
  const float* bp  = (const float*)d_in[5];
  float* out = (float*)d_out;

  const size_t BNC = (size_t)Bc * Nc * Cc;   // 4 Mi elements
  const size_t CC  = (size_t)Cc * Cc;        // 1 Mi
  char* p = (char*)d_ws;
  f16* x_h   = (f16*)p; p += BNC * sizeof(f16);
  f16* y_h   = (f16*)p; p += BNC * sizeof(f16);
  f16* Wq_h  = (f16*)p; p += CC * sizeof(f16);
  f16* Wkv_h = (f16*)p; p += 2 * CC * sizeof(f16);
  f16* Wp_h  = (f16*)p; p += CC * sizeof(f16);
  f16* qh    = (f16*)p; p += BNC * sizeof(f16);
  f16* khd   = (f16*)p; p += BNC * sizeof(f16);
  f16* vhT   = (f16*)p; p += BNC * sizeof(f16);
  f16* oh    = (f16*)p; p += BNC * sizeof(f16);

  auto cvt = [&](const float* s, f16* d, int n) {
    int blocks = (n + 255) / 256;
    if (blocks > 8192) blocks = 8192;
    cvt_f32_f16<<<blocks, 256, 0, stream>>>(s, d, n);
  };
  cvt(x,   x_h,   (int)BNC);
  cvt(y,   y_h,   (int)BNC);
  cvt(Wq,  Wq_h,  (int)CC);
  cvt(Wkv, Wkv_h, (int)(2 * CC));
  cvt(Wp,  Wp_h,  (int)CC);

  const dim3 blk(128);
  // q projection: [4096 x 1024]
  gemm_awt<<<dim3(64, 16), blk, 0, stream>>>(x_h, Wq_h, nullptr, qh, nullptr,
                                             nullptr, 0);
  // kv projection: [4096 x 2048]
  gemm_awt<<<dim3(64, 32), blk, 0, stream>>>(y_h, Wkv_h, nullptr, khd, vhT,
                                             nullptr, 1);
  // flash attention
  attn_kernel<<<dim3(Nc / 64, Bc * Hc), blk, 0, stream>>>(qh, khd, vhT, oh);
  // output projection with bias: f32 result
  gemm_awt<<<dim3(64, 16), blk, 0, stream>>>(oh, Wp_h, bp, nullptr, nullptr,
                                             out, 2);
}
